// VQVAE_61143154426546
// MI455X (gfx1250) — matrix operand order repaired and checked by reading, not compiled, not run on hardware
//
#include <hip/hip_runtime.h>
#include <hip/hip_bf16.h>
#include <math.h>

typedef __attribute__((ext_vector_type(16))) __bf16 v16bf;
typedef __attribute__((ext_vector_type(8)))  float  v8f;

__device__ __forceinline__ int iclamp(int v, int lo, int hi) {
  return v < lo ? lo : (v > hi ? hi : v);
}

// ---------------------------------------------------------------------------
// WMMA implicit-GEMM convolution.
//   out[b, co_off+m, oy, ox] (channel stride c_total)
//     = bias[m] + sum_{ci,ky,kx} w[m,ci,ky,kx] * gather(x) (+ addsrc)
// MODE 0: normal conv (stride,pad); MODE 1: ConvTranspose2d(k=4,s=2,p=1).
// MX: Cout is a multiple of 16 -> all M guards removed (straight-line stores).
// One wave computes a 16(M) x 64(N) tile; one A fragment feeds 4 WMMAs per
// K-step.  K loop = unguarded main body (K%32 bulk) + guarded tail.
// All gathers are branchless (clamped index + unconditional load + cndmask).
// ---------------------------------------------------------------------------
template<int KH, int KW, int MODE, bool MX>
__global__ __launch_bounds__(32)
void conv_wmma_kernel(const float* __restrict__ x, const float* __restrict__ w,
                      const float* __restrict__ bias, const float* __restrict__ addsrc,
                      float* __restrict__ out,
                      int Cin, int IH, int IW,
                      int Cout, int lOW, int OH,
                      int stride, int pad,
                      int co_off, int c_total)
{
  constexpr int KHW = KH * KW;
  const int lane = threadIdx.x & 31;
  const int hi = lane >> 4, lo = lane & 15;
  const int OW = 1 << lOW;
  const int OHW = OH << lOW;
  const int n_base = blockIdx.x * 64;
  const int m0 = blockIdx.y * 16;
  const int b  = blockIdx.z;
  const int K  = Cin * KHW;
  const int m  = m0 + lo;
  const bool mval = MX ? true : (m < Cout);
  const float* __restrict__ xb = x + (size_t)b * Cin * IH * IW;
  const float* __restrict__ arow = w + (size_t)(mval ? m : 0) * K;

  int oyv[4], oxv[4];
#pragma unroll
  for (int t = 0; t < 4; ++t) {
    const int n = n_base + t * 16 + lo;
    oyv[t] = n >> lOW;
    oxv[t] = n & (OW - 1);
  }

  v8f acc[4] = {};
  const int kmain = K & ~31;

  // ---------------- main loop: no K guards ----------------
  for (int k0 = 0; k0 < kmain; k0 += 32) {
    v16bf afrag;
    if (MX && ((K & 3) == 0)) {          // contiguous vector loads (ISA A runs)
      const float* ap = arow + k0 + hi * 8;
      if (k0 + 64 <= K) __builtin_prefetch(ap + 32, 0, 0);
      const float4 a0 = *(const float4*)(ap);
      const float4 a1 = *(const float4*)(ap + 4);
      const float4 a2 = *(const float4*)(ap + 16);
      const float4 a3 = *(const float4*)(ap + 20);
      afrag[0]  = (__bf16)a0.x; afrag[1]  = (__bf16)a0.y;
      afrag[2]  = (__bf16)a0.z; afrag[3]  = (__bf16)a0.w;
      afrag[4]  = (__bf16)a1.x; afrag[5]  = (__bf16)a1.y;
      afrag[6]  = (__bf16)a1.z; afrag[7]  = (__bf16)a1.w;
      afrag[8]  = (__bf16)a2.x; afrag[9]  = (__bf16)a2.y;
      afrag[10] = (__bf16)a2.z; afrag[11] = (__bf16)a2.w;
      afrag[12] = (__bf16)a3.x; afrag[13] = (__bf16)a3.y;
      afrag[14] = (__bf16)a3.z; afrag[15] = (__bf16)a3.w;
    } else {
      float av[16];
#pragma unroll
      for (int e = 0; e < 16; ++e) {
        const int kk = k0 + ((e < 8) ? e : (e + 8)) + hi * 8;
        av[e] = arow[kk];
        av[e] = mval ? av[e] : 0.f;
      }
#pragma unroll
      for (int e = 0; e < 16; ++e) afrag[e] = (__bf16)av[e];
    }

#pragma unroll
    for (int t = 0; t < 4; ++t) {
      const int oy = oyv[t], ox = oxv[t];
      float bv[16];
#pragma unroll
      for (int e = 0; e < 16; ++e) {
        const int kk = k0 + ((e < 8) ? e : (e + 8)) + hi * 8;
        const int ci = kk / KHW;             // division by compile-time constant
        const int r  = kk - ci * KHW;
        const int ky = r / KW;
        const int kx = r - ky * KW;
        bool inb;
        int iy, ix;
        if (MODE == 0) {
          iy = oy * stride - pad + ky;
          ix = ox * stride - pad + kx;
          inb = ((unsigned)iy < (unsigned)IH) && ((unsigned)ix < (unsigned)IW);
        } else {  // convT k4 s2 p1: dilated-input correlation, pad 2
          const int ty = oy + ky - 2, tx = ox + kx - 2;
          iy = ty >> 1; ix = tx >> 1;
          inb = (ty >= 0) && (tx >= 0) && !(ty & 1) && !(tx & 1) &&
                (iy < IH) && (ix < IW);
        }
        iy = iclamp(iy, 0, IH - 1);
        ix = iclamp(ix, 0, IW - 1);
        float v = xb[((size_t)ci * IH + iy) * IW + ix];
        bv[e] = inb ? v : 0.f;
      }
      v16bf bfrag;
#pragma unroll
      for (int e = 0; e < 16; ++e) bfrag[e] = (__bf16)bv[e];
      acc[t] = __builtin_amdgcn_wmma_f32_16x16x32_bf16(false, afrag, false, bfrag,
                                                       (short)0, acc[t], false, false);
    }
  }

  // ---------------- tail: guarded (only if K % 32 != 0) ----------------
  if (kmain < K) {
    const int k0 = kmain;
    float av[16];
#pragma unroll
    for (int e = 0; e < 16; ++e) {
      const int kk = k0 + ((e < 8) ? e : (e + 8)) + hi * 8;
      const bool kval = (kk < K);
      av[e] = arow[kval ? kk : 0];
      av[e] = (mval && kval) ? av[e] : 0.f;
    }
    v16bf afrag;
#pragma unroll
    for (int e = 0; e < 16; ++e) afrag[e] = (__bf16)av[e];

#pragma unroll
    for (int t = 0; t < 4; ++t) {
      const int oy = oyv[t], ox = oxv[t];
      float bv[16];
#pragma unroll
      for (int e = 0; e < 16; ++e) {
        const int kk = k0 + ((e < 8) ? e : (e + 8)) + hi * 8;
        const bool kval = (kk < K);
        const int kc = kval ? kk : 0;
        const int ci = kc / KHW;
        const int r  = kc - ci * KHW;
        const int ky = r / KW;
        const int kx = r - ky * KW;
        bool inb;
        int iy, ix;
        if (MODE == 0) {
          iy = oy * stride - pad + ky;
          ix = ox * stride - pad + kx;
          inb = ((unsigned)iy < (unsigned)IH) && ((unsigned)ix < (unsigned)IW);
        } else {
          const int ty = oy + ky - 2, tx = ox + kx - 2;
          iy = ty >> 1; ix = tx >> 1;
          inb = (ty >= 0) && (tx >= 0) && !(ty & 1) && !(tx & 1) &&
                (iy < IH) && (ix < IW);
        }
        iy = iclamp(iy, 0, IH - 1);
        ix = iclamp(ix, 0, IW - 1);
        float v = xb[((size_t)ci * IH + iy) * IW + ix];
        bv[e] = (kval && inb) ? v : 0.f;
      }
      v16bf bfrag;
#pragma unroll
      for (int e = 0; e < 16; ++e) bfrag[e] = (__bf16)bv[e];
      acc[t] = __builtin_amdgcn_wmma_f32_16x16x32_bf16(false, afrag, false, bfrag,
                                                       (short)0, acc[t], false, false);
    }
  }

  // ---------------- epilogue ----------------
  float br[8];
#pragma unroll
  for (int r = 0; r < 8; ++r) {
    const int mm = m0 + r + hi * 8;
    const bool ok = MX || (mm < Cout);
    br[r] = bias ? bias[ok ? mm : 0] : 0.f;
    br[r] = ok ? br[r] : 0.f;
  }
#pragma unroll
  for (int t = 0; t < 4; ++t) {
    const int n = n_base + t * 16 + lo;
    const size_t base = (size_t)(b * c_total + co_off + m0 + hi * 8) * OHW + n;
#pragma unroll
    for (int r = 0; r < 8; ++r) {
      if (MX || (m0 + r + hi * 8) < Cout) {
        const size_t oidx = base + (size_t)r * OHW;
        float v = acc[t][r] + br[r];
        if (addsrc) v += addsrc[oidx];
        out[oidx] = v;
      }
    }
  }
}

// ---------------------------------------------------------------------------
// Strided batched GEMM via WMMA: C = alpha * A * B  (per blockIdx.z batch).
// All shapes at the call sites are exact tile multiples -> no guards at all.
// ---------------------------------------------------------------------------
__global__ __launch_bounds__(32)
void gemm_wmma_kernel(const float* __restrict__ A, const float* __restrict__ Bm,
                      float* __restrict__ C,
                      int K,
                      int sa_m, int sa_k, long a_bs,
                      int sb_k, int sb_n, long b_bs,
                      int sc_m, int sc_n, long c_bs,
                      float alpha)
{
  const int lane = threadIdx.x & 31;
  const int hi = lane >> 4, lo = lane & 15;
  const int n0 = blockIdx.x * 16, m0 = blockIdx.y * 16;
  const long za = (long)blockIdx.z * a_bs + (long)(m0 + lo) * sa_m;
  const long zb = (long)blockIdx.z * b_bs + (long)(n0 + lo) * sb_n;
  const long zc = (long)blockIdx.z * c_bs;

  v8f acc = {};
  for (int k0 = 0; k0 < K; k0 += 32) {
    float av[16], bv[16];
#pragma unroll
    for (int e = 0; e < 16; ++e) {
      const int kk = k0 + ((e < 8) ? e : (e + 8)) + hi * 8;
      av[e] = A[za + (long)kk * sa_k];
    }
#pragma unroll
    for (int e = 0; e < 16; ++e) {
      const int kk = k0 + ((e < 8) ? e : (e + 8)) + hi * 8;
      bv[e] = Bm[zb + (long)kk * sb_k];
    }
    v16bf afrag, bfrag;
#pragma unroll
    for (int e = 0; e < 16; ++e) { afrag[e] = (__bf16)av[e]; bfrag[e] = (__bf16)bv[e]; }
    acc = __builtin_amdgcn_wmma_f32_16x16x32_bf16(false, afrag, false, bfrag,
                                                  (short)0, acc, false, false);
  }
  const long cb = zc + (long)(m0 + hi * 8) * sc_m + (long)(n0 + lo) * sc_n;
#pragma unroll
  for (int r = 0; r < 8; ++r)
    C[cb + (long)r * sc_m] = alpha * acc[r];
}

// ---------------------------------------------------------------------------
// GroupNorm (+optional SiLU). grid = (32 groups, B); one block per (b,group).
// ---------------------------------------------------------------------------
__global__ __launch_bounds__(256)
void gn_kernel(const float* __restrict__ x, const float* __restrict__ g,
               const float* __restrict__ beta, float* __restrict__ out,
               int C, int HW, int cg, int act)
{
  __shared__ float s0[256], s1[256];
  const int grp = blockIdx.x, b = blockIdx.y;
  const int tid = threadIdx.x;
  const int nel = cg * HW;
  const size_t base = ((size_t)b * C + (size_t)grp * cg) * HW;
  float sum = 0.f, sq = 0.f;
  for (int j = tid; j < nel; j += 256) { float v = x[base + j]; sum += v; sq += v * v; }
  s0[tid] = sum; s1[tid] = sq;
  __syncthreads();
  for (int s = 128; s > 0; s >>= 1) {
    if (tid < s) { s0[tid] += s0[tid + s]; s1[tid] += s1[tid + s]; }
    __syncthreads();
  }
  const float mean = s0[0] / (float)nel;
  const float var  = s1[0] / (float)nel - mean * mean;
  const float rstd = rsqrtf(var + 1e-5f);
  for (int j = tid; j < nel; j += 256) {
    const int c = grp * cg + j / HW;
    float v = (x[base + j] - mean) * rstd * g[c] + beta[c];
    if (act) v = v / (1.f + __expf(-v));   // SiLU
    out[base + j] = v;
  }
}

__global__ void avgpool2_kernel(const float* __restrict__ x, float* __restrict__ out,
                                int H, int W, int total)
{
  int i = blockIdx.x * blockDim.x + threadIdx.x;
  if (i >= total) return;
  const int OW = W >> 1, OH = H >> 1;
  int ox = i & (OW - 1); int t = i >> (31 - __clz(OW));
  int oy = t & (OH - 1); t >>= (31 - __clz(OH));        // t = b*C + c
  const float* p = x + ((size_t)t * H + (size_t)oy * 2) * W + ox * 2;
  out[i] = 0.25f * (p[0] + p[1] + p[W] + p[W + 1]);
}

__global__ void upsample2_kernel(const float* __restrict__ x, float* __restrict__ out,
                                 int H, int W, int total)
{
  int i = blockIdx.x * blockDim.x + threadIdx.x;
  if (i >= total) return;
  const int OW = W * 2, OH = H * 2;
  int ox = i & (OW - 1); int t = i >> (31 - __clz(OW));
  int oy = t & (OH - 1); t >>= (31 - __clz(OH));
  float sy = (oy + 0.5f) * 0.5f - 0.5f;      // half-pixel, align_corners=False
  float sx = (ox + 0.5f) * 0.5f - 0.5f;
  int y0 = (int)floorf(sy), x0 = (int)floorf(sx);
  float fy = sy - y0, fx = sx - x0;
  int y1 = y0 + 1, x1 = x0 + 1;
  y0 = iclamp(y0, 0, H - 1); y1 = iclamp(y1, 0, H - 1);
  x0 = iclamp(x0, 0, W - 1); x1 = iclamp(x1, 0, W - 1);
  const float* p = x + (size_t)t * H * W;
  out[i] = (1.f - fy) * ((1.f - fx) * p[y0 * W + x0] + fx * p[y0 * W + x1])
         +        fy  * ((1.f - fx) * p[y1 * W + x0] + fx * p[y1 * W + x1]);
}

// softmax over contiguous rows of length 256 (attention at 16x16)
__global__ __launch_bounds__(256)
void softmax256_kernel(float* __restrict__ s)
{
  __shared__ float red[256];
  const int tid = threadIdx.x;
  float* row = s + (size_t)blockIdx.x * 256;
  float v = row[tid];
  red[tid] = v; __syncthreads();
  for (int k = 128; k > 0; k >>= 1) {
    if (tid < k) red[tid] = fmaxf(red[tid], red[tid + k]);
    __syncthreads();
  }
  const float mx = red[0]; __syncthreads();
  float e = __expf(v - mx);
  red[tid] = e; __syncthreads();
  for (int k = 128; k > 0; k >>= 1) {
    if (tid < k) red[tid] += red[tid + k];
    __syncthreads();
  }
  row[tid] = e / red[0];
}

// ------------------------------- VQ loss -----------------------------------
__global__ void vq_zero_kernel(float* acc) { *acc = 0.f; }

__global__ __launch_bounds__(256)
void vq_kernel(const float* __restrict__ z, const float* __restrict__ cb,
               float* __restrict__ acc, int P, int NE)
{
  __shared__ float sd[256];
  __shared__ int   si[256];
  const int tid = threadIdx.x;
  const int vec = blockIdx.x;            // b*P + p
  const int b = vec / P, p = vec - (vec / P) * P;
  float xv[4]; float xn = 0.f;
#pragma unroll
  for (int c = 0; c < 4; ++c) { xv[c] = z[((size_t)b * 4 + c) * P + p]; xn += xv[c] * xv[c]; }
  float best = 3.4e38f; int bidx = 0x7fffffff;
  for (int e = tid; e < NE; e += 256) {
    const float* ce = cb + (size_t)e * 4;
    float dot = 0.f, cn = 0.f;
#pragma unroll
    for (int c = 0; c < 4; ++c) { dot += xv[c] * ce[c]; cn += ce[c] * ce[c]; }
    float d = xn + cn - 2.f * dot;
    if (d < best) { best = d; bidx = e; }
  }
  sd[tid] = best; si[tid] = bidx; __syncthreads();
  for (int k = 128; k > 0; k >>= 1) {
    if (tid < k) {
      if (sd[tid + k] < sd[tid] || (sd[tid + k] == sd[tid] && si[tid + k] < si[tid])) {
        sd[tid] = sd[tid + k]; si[tid] = si[tid + k];
      }
    }
    __syncthreads();
  }
  if (tid == 0) {
    const float* ce = cb + (size_t)si[0] * 4;
    float s = 0.f;
#pragma unroll
    for (int c = 0; c < 4; ++c) { float d = ce[c] - xv[c]; s += d * d; }
    atomicAdd(acc, s);
  }
}

__global__ void vq_final_kernel(const float* acc, float* out, float scale)
{ out[0] = acc[0] * scale; }

// ===========================================================================
// Host orchestration
// ===========================================================================
namespace {

constexpr int NB = 2;  // batch

struct Cursor {
  void* const* in; int i;
  const float* nx() { return (const float*)in[i++]; }
};

inline int ilog2(int v) { int l = 0; while ((1 << l) < v) ++l; return l; }

inline void launch_conv(hipStream_t st, const float* x, const float* w, const float* b,
                        const float* add, float* out, int Cin, int IH, int IW,
                        int Cout, int OH, int OW, int kh, int kw, int s, int p,
                        int co_off, int ctot, int mode)
{
  (void)kw;
  const int lOW = ilog2(OW);
  const bool mx = (Cout % 16 == 0);
  dim3 grid((OH * OW + 63) / 64, (Cout + 15) / 16, NB);
#define CONV_LAUNCH(A, B, C)                                                     \
  do {                                                                           \
    if (mx) conv_wmma_kernel<A, B, C, true><<<grid, 32, 0, st>>>(                \
        x, w, b, add, out, Cin, IH, IW, Cout, lOW, OH, s, p, co_off, ctot);      \
    else    conv_wmma_kernel<A, B, C, false><<<grid, 32, 0, st>>>(               \
        x, w, b, add, out, Cin, IH, IW, Cout, lOW, OH, s, p, co_off, ctot);      \
  } while (0)
  if (mode == 1)      CONV_LAUNCH(4, 4, 1);
  else if (kh == 1)   CONV_LAUNCH(1, 1, 0);
  else if (kh == 3)   CONV_LAUNCH(3, 3, 0);
  else                CONV_LAUNCH(4, 4, 0);
#undef CONV_LAUNCH
}

inline void launch_gn(hipStream_t st, const float* x, const float* g, const float* b,
                      float* out, int C, int HW, int act)
{
  dim3 grid(32, NB);
  gn_kernel<<<grid, 256, 0, st>>>(x, g, b, out, C, HW, C / 32, act);
}

void res_block(hipStream_t st, Cursor& cu, const float* x, float* out,
               float* g, float* h, float* sc, int cin, int cout, int H)
{
  const float* g0g = cu.nx(); const float* g0b = cu.nx();
  const float* w0  = cu.nx(); const float* b0  = cu.nx();
  const float* g1g = cu.nx(); const float* g1b = cu.nx();
  const float* w1  = cu.nx(); const float* b1  = cu.nx();
  const float* scw = nullptr; const float* scb = nullptr;
  if (cin != cout) { scw = cu.nx(); scb = cu.nx(); }
  const int HW = H * H;
  launch_gn(st, x, g0g, g0b, g, cin, HW, 1);
  launch_conv(st, g, w0, b0, nullptr, h, cin, H, H, cout, H, H, 3, 3, 1, 1, 0, cout, 0);
  launch_gn(st, h, g1g, g1b, g, cout, HW, 1);
  const float* add = x;
  if (scw) {
    launch_conv(st, x, scw, scb, nullptr, sc, cin, H, H, cout, H, H, 1, 1, 1, 0, 0, cout, 0);
    add = sc;
  }
  launch_conv(st, g, w1, b1, add, out, cout, H, H, cout, H, H, 3, 3, 1, 1, 0, cout, 0);
}

void attn_block(hipStream_t st, Cursor& cu, const float* x, float* out, float* g,
                float* q, float* k, float* v, float* s, float* o, int C, int H)
{
  const float* gg = cu.nx(); const float* gb = cu.nx();
  const float* wq = cu.nx(); const float* bq = cu.nx();
  const float* wk = cu.nx(); const float* bk = cu.nx();
  const float* wv = cu.nx(); const float* bv = cu.nx();
  const float* wo = cu.nx(); const float* bo = cu.nx();
  const int P = H * H, heads = 4, dh = C / heads;
  launch_gn(st, x, gg, gb, g, C, P, 0);
  launch_conv(st, g, wq, bq, nullptr, q, C, H, H, C, H, H, 1, 1, 1, 0, 0, C, 0);
  launch_conv(st, g, wk, bk, nullptr, k, C, H, H, C, H, H, 1, 1, 1, 0, 0, C, 0);
  launch_conv(st, g, wv, bv, nullptr, v, C, H, H, C, H, H, 1, 1, 1, 0, 0, C, 0);
  // S[p,q'] = sum_d Q[p,d] K[q',d] / sqrt(dh)   (q/k stored as (z, d, p))
  {
    dim3 grid(P / 16, P / 16, NB * heads);
    gemm_wmma_kernel<<<grid, 32, 0, st>>>(q, k, s, dh,
        1, P, (long)dh * P,
        P, 1, (long)dh * P,
        P, 1, (long)P * P,
        1.f / sqrtf((float)dh));
  }
  softmax256_kernel<<<NB * heads * P, 256, 0, st>>>(s);
  // O[p,d] = sum_q' S[p,q'] V[q',d]; store as (z, d, p)
  {
    dim3 grid(dh / 16, P / 16, NB * heads);
    gemm_wmma_kernel<<<grid, 32, 0, st>>>(s, v, o, P,
        P, 1, (long)P * P,
        1, P, (long)dh * P,
        1, P, (long)dh * P,
        1.f);
  }
  launch_conv(st, o, wo, bo, x, out, C, H, H, C, H, H, 1, 1, 1, 0, 0, C, 0);
}

void down_block(hipStream_t st, Cursor& cu, const float* x, float* out, float* tmp,
                int cin, int cout, int H)
{
  const float* cw = cu.nx(); const float* cb = cu.nx();
  const float* pw = cu.nx(); const float* pb = cu.nx();
  const int OHd = H / 2, half = cout / 2;
  launch_conv(st, x, cw, cb, nullptr, out, cin, H, H, half, OHd, OHd, 4, 4, 2, 1, 0, cout, 0);
  const int total = NB * cin * OHd * OHd;
  avgpool2_kernel<<<(total + 255) / 256, 256, 0, st>>>(x, tmp, H, H, total);
  launch_conv(st, tmp, pw, pb, nullptr, out, cin, OHd, OHd, half, OHd, OHd, 1, 1, 1, 0, half, cout, 0);
}

void up_block(hipStream_t st, Cursor& cu, const float* x, float* out, float* tmp,
              int cin, int H)
{
  const float* tw = cu.nx(); const float* tb = cu.nx();
  const float* pw = cu.nx(); const float* pb = cu.nx();
  const int OHu = H * 2, half = cin / 2;
  launch_conv(st, x, tw, tb, nullptr, out, cin, H, H, half, OHu, OHu, 4, 4, 2, 1, 0, cin, 1);
  const int total = NB * cin * OHu * OHu;
  upsample2_kernel<<<(total + 255) / 256, 256, 0, st>>>(x, tmp, H, H, total);
  launch_conv(st, tmp, pw, pb, nullptr, out, cin, OHu, OHu, half, OHu, OHu, 1, 1, 1, 0, half, cin, 0);
}

}  // namespace

extern "C" void kernel_launch(void* const* d_in, const int* in_sizes, int n_in,
                              void* d_out, int out_size, void* d_ws, size_t ws_size,
                              hipStream_t stream)
{
  (void)in_sizes; (void)n_in; (void)out_size; (void)ws_size;

  // Inputs flattened in setup_inputs()/build_params() insertion order.
  Cursor cu{d_in, 0};
  const float* x_in = cu.nx();

  // Workspace: 5 big slots (max activation = 2*128*128*128 floats) + attn region.
  const size_t SLOT = 4194304;
  float* ws = (float*)d_ws;
  float* S0 = ws;
  float* S1 = ws + 1 * SLOT;
  float* SG = ws + 2 * SLOT;   // groupnorm / pooled / upsampled scratch
  float* SH = ws + 3 * SLOT;   // res-block intermediate
  float* SC = ws + 4 * SLOT;   // shortcut
  float* qb = ws + 5 * SLOT;
  float* kb = qb + 131072;
  float* vb = kb + 131072;
  float* ob = vb + 131072;
  float* sb = ob + 131072;     // attention scores (2MB)
  float* acc = sb + 524288;

  float* cur = S0; float* nxt = S1;
  auto swp = [&]() { float* t = cur; cur = nxt; nxt = t; };

  // ---- encoder ----
  { const float* w = cu.nx(); const float* b = cu.nx();          // enc_conv_in
    launch_conv(stream, x_in, w, b, nullptr, cur, 3, 128, 128, 64, 128, 128, 3, 3, 1, 1, 0, 64, 0); }

  int H = 128;
  for (int lvl = 0; lvl <= 3; ++lvl) {
    const int cin  = 64 * ((lvl + 1 < 4) ? lvl + 1 : 4);
    const int cout = 64 * ((lvl + 2 < 4) ? lvl + 2 : 4);
    for (int i = 0; i < 2; ++i) {
      const int ci = (i == 0) ? cin : cout;
      res_block(stream, cu, cur, nxt, SG, SH, SC, ci, cout, H); swp();
      if (H <= 16) { attn_block(stream, cu, cur, nxt, SG, qb, kb, vb, sb, ob, cout, H); swp(); }
    }
    if (lvl != 3) { down_block(stream, cu, cur, nxt, SG, cout, cout, H); H /= 2; swp(); }
  }

  // enc_mid
  res_block(stream, cu, cur, nxt, SG, SH, SC, 256, 256, 16); swp();
  attn_block(stream, cu, cur, nxt, SG, qb, kb, vb, sb, ob, 256, 16); swp();
  res_block(stream, cu, cur, nxt, SG, SH, SC, 256, 256, 16); swp();

  // enc_conv_out: GN+SiLU + conv3x3 256->4
  { const float* gg = cu.nx(); const float* gb = cu.nx();
    const float* w  = cu.nx(); const float* b  = cu.nx();
    launch_gn(stream, cur, gg, gb, SG, 256, 256, 1);
    launch_conv(stream, SG, w, b, nullptr, nxt, 256, 16, 16, 4, 16, 16, 3, 3, 1, 1, 0, 4, 0);
    swp(); }

  // pre_q 1x1 4->4
  { const float* w = cu.nx(); const float* b = cu.nx();
    launch_conv(stream, cur, w, b, nullptr, nxt, 4, 16, 16, 4, 16, 16, 1, 1, 1, 0, 0, 4, 0);
    swp(); }

  // quantize: loss only (forward output is the identity pass-through of z)
  { const float* cb = cu.nx();                                    // codebook
    vq_zero_kernel<<<1, 1, 0, stream>>>(acc);
    vq_kernel<<<NB * 256, 256, 0, stream>>>(cur, cb, acc, 256, 8192);
    vq_final_kernel<<<1, 1, 0, stream>>>(acc, (float*)d_out + 98304, 1.25f / 2048.f); }

  // post_q 1x1 4->4
  { const float* w = cu.nx(); const float* b = cu.nx();
    launch_conv(stream, cur, w, b, nullptr, nxt, 4, 16, 16, 4, 16, 16, 1, 1, 1, 0, 0, 4, 0);
    swp(); }

  // dec_conv_in 3x3 4->256
  { const float* w = cu.nx(); const float* b = cu.nx();
    launch_conv(stream, cur, w, b, nullptr, nxt, 4, 16, 16, 256, 16, 16, 3, 3, 1, 1, 0, 256, 0);
    swp(); }

  // dec_mid
  res_block(stream, cu, cur, nxt, SG, SH, SC, 256, 256, 16); swp();
  attn_block(stream, cu, cur, nxt, SG, qb, kb, vb, sb, ob, 256, 16); swp();
  res_block(stream, cu, cur, nxt, SG, SH, SC, 256, 256, 16); swp();

  // ---- decoder ----
  H = 16; int block_in = 256;
  for (int lvl = 3; lvl >= 0; --lvl) {
    const int cout = 64 * ((lvl + 1 < 4) ? lvl + 1 : 4);
    if (lvl != 3) { up_block(stream, cu, cur, nxt, SG, block_in, H); H *= 2; swp(); }
    for (int i = 0; i < 2; ++i) {
      const int ci = (i == 0) ? block_in : cout;
      res_block(stream, cu, cur, nxt, SG, SH, SC, ci, cout, H); swp();
      if (H <= 16) { attn_block(stream, cu, cur, nxt, SG, qb, kb, vb, sb, ob, cout, H); swp(); }
    }
    block_in = cout;
  }

  // dec_conv_out: GN (no act) + conv3x3 64->3 straight into d_out
  { const float* gg = cu.nx(); const float* gb = cu.nx();
    const float* w  = cu.nx(); const float* b  = cu.nx();
    launch_gn(stream, cur, gg, gb, SG, 64, 128 * 128, 0);
    launch_conv(stream, SG, w, b, nullptr, (float*)d_out, 64, 128, 128, 3, 128, 128, 3, 3, 1, 1, 0, 3, 0); }
}
